// MultiHeadAttention_72524817760467
// MI455X (gfx1250) — matrix operand order repaired and checked by reading, not compiled, not run on hardware
//
#include <hip/hip_runtime.h>

typedef __attribute__((ext_vector_type(16))) __bf16 v16bf;
typedef __attribute__((ext_vector_type(8)))  float  v8f;

#define HIDDEN 1024
#define HEADS  16
#define HD     64
#define BATCH  4
#define SEQ    2048
#define MTOT   (BATCH * SEQ)   // 8192 rows

union AFrag { v16bf v; unsigned u[8]; };

__device__ __forceinline__ v8f wmma_bf16(v16bf a, v16bf b, v8f c) {
  // D = A(16x32) x B(32x16) + C, f32 accumulate
  return __builtin_amdgcn_wmma_f32_16x16x32_bf16(false, a, false, b, (short)0, c, false, false);
}

// Low 32 bits of a generic pointer to __shared__ = addrspace(3) byte offset
__device__ __forceinline__ unsigned lds_off(const void* p) {
  return (unsigned)(unsigned long long)p;
}

// Async global -> LDS copy, 16B per lane, tracked by ASYNCcnt (CDNA5 path)
__device__ __forceinline__ void async_b128(unsigned lds_addr, const void* gptr) {
  asm volatile("global_load_async_to_lds_b128 %0, %1, off"
               :: "v"(lds_addr), "v"(gptr) : "memory");
}

__device__ __forceinline__ void wait_async0() {
  asm volatile("s_wait_asynccnt 0" ::: "memory");
}

// ---------------- fp32 -> bf16 conversion (4 elems / thread) ----------------
__global__ __launch_bounds__(256) void cvt_kernel(const float* __restrict__ src,
                                                  void* __restrict__ dstv, int n4) {
  int i = blockIdx.x * blockDim.x + threadIdx.x;
  if (i >= n4) return;
  const float4 f = reinterpret_cast<const float4*>(src)[i];
  union { __bf16 h[4]; uint2 q; } u;
  u.h[0] = (__bf16)f.x; u.h[1] = (__bf16)f.y;
  u.h[2] = (__bf16)f.z; u.h[3] = (__bf16)f.w;
  reinterpret_cast<uint2*>(dstv)[i] = u.q;
}

// ---------------- GEMM: out[m][n] = A[m][:] . W[n][:] + bias[n] ----------------
// Double-buffered async-LDS pipeline. MODE 0: bf16 [b][h][s][d] QKV layout.
// MODE 1: fp32 row-major.
template<int MODE>
__global__ __launch_bounds__(256) void gemm_kernel(const void* __restrict__ Av,
                                                   const void* __restrict__ Wtv,
                                                   const float* __restrict__ bias,
                                                   void* __restrict__ outp) {
  const __bf16* __restrict__ A  = (const __bf16*)Av;
  const __bf16* __restrict__ Wt = (const __bf16*)Wtv;
  __shared__ __align__(16) __bf16 Al[2][128 * 32];  // 2 x 8 KB
  __shared__ __align__(16) __bf16 Bl[2][64 * 32];   // 2 x 4 KB

  const int t = threadIdx.x, w = t >> 5, l = t & 31, half = l >> 4, ln = l & 15;
  const int m0 = blockIdx.x * 128, n0 = blockIdx.y * 64;

  const int ea = t * 16, ra = ea >> 5, ca = ea & 31;  // A tile: 2 x b128 / thread
  const int eb = t * 8,  rb = eb >> 5, cb = eb & 31;  // B tile: 1 x b128 / thread
  const __bf16* gA = &A [(m0 + ra) * HIDDEN + ca];
  const __bf16* gB = &Wt[(n0 + rb) * HIDDEN + cb];

  // prologue: issue k0 = 0 into buffer 0
  async_b128(lds_off(&Al[0][ra * 32 + ca]),     gA);
  async_b128(lds_off(&Al[0][ra * 32 + ca + 8]), gA + 8);
  async_b128(lds_off(&Bl[0][rb * 32 + cb]),     gB);

  const v8f vz = {0.f,0.f,0.f,0.f,0.f,0.f,0.f,0.f};
  v8f acc[4]; acc[0]=vz; acc[1]=vz; acc[2]=vz; acc[3]=vz;

  int cur = 0;
  for (int k0 = 0; k0 < HIDDEN; k0 += 32) {
    wait_async0();        // own async copies landed in LDS
    __syncthreads();      // all waves' copies visible; prior buf reads done
    if (k0 + 32 < HIDDEN) {  // issue next tile into the other buffer
      const int nb = cur ^ 1, kn = k0 + 32;
      async_b128(lds_off(&Al[nb][ra * 32 + ca]),     gA + kn);
      async_b128(lds_off(&Al[nb][ra * 32 + ca + 8]), gA + kn + 8);
      async_b128(lds_off(&Bl[nb][rb * 32 + cb]),     gB + kn);
    }

    AFrag af;
#pragma unroll
    for (int v = 0; v < 8; ++v) {
      const int kb = (half << 4) + (v << 1); // contiguous k = 16*half + s
      af.u[v] = *reinterpret_cast<const unsigned*>(&Al[cur][(w * 16 + ln) * 32 + kb]);
    }
#pragma unroll
    for (int j = 0; j < 4; ++j) {
      AFrag bf;
#pragma unroll
      for (int v = 0; v < 8; ++v)                                  // contiguous k = 16*half + s, as A
        bf.u[v] = *reinterpret_cast<const unsigned*>(&Bl[cur][(j * 16 + ln) * 32 + (half << 4) + (v << 1)]);
      acc[j] = wmma_bf16(af.v, bf.v, acc[j]);
    }
    cur ^= 1;
  }

#pragma unroll
  for (int j = 0; j < 4; ++j) {
    const int n = n0 + j * 16 + ln;
    const float bn = bias[n];
#pragma unroll
    for (int v = 0; v < 8; ++v) {
      const int m = m0 + w * 16 + (half << 3) + v;   // ISA C/D layout row
      const float val = acc[j][v] + bn;
      if (MODE == 0) {
        const int bi = m >> 11, s = m & (SEQ - 1);
        const int hh = n >> 6,  d = n & 63;
        reinterpret_cast<__bf16*>(outp)[(((bi * HEADS + hh) * SEQ) + s) * HD + d] = (__bf16)val;
      } else {
        reinterpret_cast<float*>(outp)[m * HIDDEN + n] = val;
      }
    }
  }
}

// ---------------- Flash attention: one (b,h) per block, 128 q-rows ----------------
__global__ __launch_bounds__(256) void attn_kernel(const void* __restrict__ Qv,
                                                   const void* __restrict__ Kv,
                                                   const void* __restrict__ Vv,
                                                   const float* __restrict__ head_mask,
                                                   void* __restrict__ ctxv) {
  const __bf16* __restrict__ Q = (const __bf16*)Qv;
  const __bf16* __restrict__ K = (const __bf16*)Kv;
  const __bf16* __restrict__ V = (const __bf16*)Vv;
  __bf16* __restrict__ ctx = (__bf16*)ctxv;

  __shared__ __align__(16) __bf16 Kl[32 * 64];      // [key][d]
  __shared__ __align__(16) __bf16 Vt[64 * 32];      // [d][key] (transposed)
  __shared__ __align__(16) __bf16 Pl[8 * 16 * 32];  // per-wave P patch

  const int bh = blockIdx.x, b = bh >> 4, h = bh & 15;
  const int t = threadIdx.x, w = t >> 5, l = t & 31, half = l >> 4, ln = l & 15;
  const int q0 = blockIdx.y * 128 + w * 16;
  const int base = bh * SEQ * HD;

  // Q fragments (d=0..31, 32..63), register-resident for the whole kernel
  AFrag qf[2];
  {
    const int row = base + (q0 + ln) * HD;
#pragma unroll
    for (int f = 0; f < 2; ++f)
#pragma unroll
      for (int v = 0; v < 8; ++v) {
        const int kb = f * 32 + (half << 4) + (v << 1);
        qf[f].u[v] = *reinterpret_cast<const unsigned*>(&Q[row + kb]);
      }
  }

  const v8f vz = {0.f,0.f,0.f,0.f,0.f,0.f,0.f,0.f};
  v8f o[4]; o[0]=vz; o[1]=vz; o[2]=vz; o[3]=vz;
  float mst[8], lst[8];
#pragma unroll
  for (int v = 0; v < 8; ++v) { mst[v] = -3.402823466e38f; lst[v] = 0.f; }

  const int er = (t * 8) >> 6, ec = (t * 8) & 63;   // cooperative tile load coords

  for (int kb0 = 0; kb0 < SEQ; kb0 += 32) {
    __syncthreads();
    uint4 kq = *reinterpret_cast<const uint4*>(&K[base + (kb0 + er) * HD + ec]);
    union { uint4 q; __bf16 hh[8]; } vq;
    vq.q = *reinterpret_cast<const uint4*>(&V[base + (kb0 + er) * HD + ec]);
    *reinterpret_cast<uint4*>(&Kl[er * 64 + ec]) = kq;
#pragma unroll
    for (int i = 0; i < 8; ++i) Vt[(ec + i) * 32 + er] = vq.hh[i];  // transpose
    __syncthreads();

    // S = Q . K^T over d=64 (two WMMA K-steps), two 16-key n-tiles
    v8f sc[2];
#pragma unroll
    for (int j = 0; j < 2; ++j) {
      AFrag kf0, kf1;
#pragma unroll
      for (int v = 0; v < 8; ++v) {
        kf0.u[v] = *reinterpret_cast<const unsigned*>(&Kl[(j * 16 + ln) * 64 + (half << 4) + (v << 1)]);
        kf1.u[v] = *reinterpret_cast<const unsigned*>(&Kl[(j * 16 + ln) * 64 + 32 + (half << 4) + (v << 1)]);
      }
      v8f c0 = wmma_bf16(qf[0].v, kf0.v, vz);
      sc[j]  = wmma_bf16(qf[1].v, kf1.v, c0);
    }

    // online softmax (rows live in half-wave, reduce with shfl_xor 8/4/2/1)
    float p0[8], p1[8];
#pragma unroll
    for (int v = 0; v < 8; ++v) {
      float s0 = sc[0][v] * 0.125f, s1 = sc[1][v] * 0.125f;  // 1/sqrt(64)
      float rm = fmaxf(s0, s1);
      rm = fmaxf(rm, __shfl_xor(rm, 8, 32));
      rm = fmaxf(rm, __shfl_xor(rm, 4, 32));
      rm = fmaxf(rm, __shfl_xor(rm, 2, 32));
      rm = fmaxf(rm, __shfl_xor(rm, 1, 32));
      const float mn = fmaxf(mst[v], rm);
      const float al = __expf(mst[v] - mn);
      const float e0 = __expf(s0 - mn), e1 = __expf(s1 - mn);
      float rs = e0 + e1;
      rs += __shfl_xor(rs, 8, 32);
      rs += __shfl_xor(rs, 4, 32);
      rs += __shfl_xor(rs, 2, 32);
      rs += __shfl_xor(rs, 1, 32);
      lst[v] = lst[v] * al + rs;
      mst[v] = mn;
      p0[v] = e0; p1[v] = e1;
      o[0][v] *= al; o[1][v] *= al; o[2][v] *= al; o[3][v] *= al;
    }

    // C-layout -> A-layout for P via this wave's private LDS patch
    // (same-wave DS ops execute in order: store->load is safe without barrier)
#pragma unroll
    for (int v = 0; v < 8; ++v) {
      __bf16* pr = &Pl[w * 512 + ((half << 3) + v) * 32];
      pr[ln]      = (__bf16)p0[v];
      pr[16 + ln] = (__bf16)p1[v];
    }
    AFrag pf;
#pragma unroll
    for (int v = 0; v < 8; ++v) {
      const int kb = (half << 4) + (v << 1);
      pf.u[v] = *reinterpret_cast<const unsigned*>(&Pl[w * 512 + ln * 32 + kb]);
    }

    // O += P @ V  (alpha-rescale already folded into C operand)
#pragma unroll
    for (int tt = 0; tt < 4; ++tt) {
      AFrag vf;
#pragma unroll
      for (int v = 0; v < 8; ++v)
        vf.u[v] = *reinterpret_cast<const unsigned*>(&Vt[(tt * 16 + ln) * 32 + (half << 4) + (v << 1)]);
      o[tt] = wmma_bf16(pf.v, vf.v, o[tt]);
    }
  }

  // epilogue: divide by l, apply post-softmax head mask (linear, so exact)
  const float hm = head_mask[h];
#pragma unroll
  for (int v = 0; v < 8; ++v) {
    const float inv = hm / lst[v];
    const int s = q0 + (half << 3) + v;
#pragma unroll
    for (int tt = 0; tt < 4; ++tt) {
      const int d = tt * 16 + ln;
      ctx[(b * SEQ + s) * HIDDEN + h * HD + d] = (__bf16)(o[tt][v] * inv);
    }
  }
}

// ---------------- launch ----------------
extern "C" void kernel_launch(void* const* d_in, const int* in_sizes, int n_in,
                              void* d_out, int out_size, void* d_ws, size_t ws_size,
                              hipStream_t stream) {
  (void)in_sizes; (void)n_in; (void)out_size; (void)ws_size;
  const float* x  = (const float*)d_in[0];
  const float* Wq = (const float*)d_in[1];
  const float* bq = (const float*)d_in[2];
  const float* Wk = (const float*)d_in[3];
  const float* bk = (const float*)d_in[4];
  const float* Wv = (const float*)d_in[5];
  const float* bv = (const float*)d_in[6];
  const float* Wo = (const float*)d_in[7];
  const float* bo = (const float*)d_in[8];
  const float* hm = (const float*)d_in[9];

  char* ws = (char*)d_ws;
  const size_t MB = 1u << 20;
  void* xb  = (void*)(ws +  0 * MB);  // 16 MB  x bf16
  void* Wqb = (void*)(ws + 16 * MB);  //  2 MB
  void* Wkb = (void*)(ws + 18 * MB);
  void* Wvb = (void*)(ws + 20 * MB);
  void* Wob = (void*)(ws + 22 * MB);
  void* Qd  = (void*)(ws + 24 * MB);  // 16 MB  [b][h][s][d] bf16
  void* Kd  = (void*)(ws + 40 * MB);
  void* Vd  = (void*)(ws + 56 * MB);
  void* Cd  = (void*)(ws + 72 * MB);  // 16 MB  ctx bf16 row-major

  const int nx4 = MTOT * HIDDEN / 4;     // 2,097,152
  const int nw4 = HIDDEN * HIDDEN / 4;   //   262,144
  cvt_kernel<<<nx4 / 256, 256, 0, stream>>>(x,  xb,  nx4);
  cvt_kernel<<<nw4 / 256, 256, 0, stream>>>(Wq, Wqb, nw4);
  cvt_kernel<<<nw4 / 256, 256, 0, stream>>>(Wk, Wkb, nw4);
  cvt_kernel<<<nw4 / 256, 256, 0, stream>>>(Wv, Wvb, nw4);
  cvt_kernel<<<nw4 / 256, 256, 0, stream>>>(Wo, Wob, nw4);

  dim3 gg(MTOT / 128, HIDDEN / 64);      // (64, 16)
  gemm_kernel<0><<<gg, 256, 0, stream>>>(xb, Wqb, bq, Qd);
  gemm_kernel<0><<<gg, 256, 0, stream>>>(xb, Wkb, bk, Kd);
  gemm_kernel<0><<<gg, 256, 0, stream>>>(xb, Wvb, bv, Vd);

  attn_kernel<<<dim3(BATCH * HEADS, SEQ / 128), 256, 0, stream>>>(Qd, Kd, Vd, hm, Cd);

  gemm_kernel<1><<<gg, 256, 0, stream>>>(Cd, Wob, bo, d_out);
}
